// EncoderRNN_68083821576341
// MI455X (gfx1250) — compile-verified, matrix operand order checked by reading, not run on hardware
//
#include <hip/hip_runtime.h>
#include <hip/hip_bf16.h>

// ---------------------------------------------------------------------------
// GRU (2-layer, shared weights) for B=32, T=2048, H=256 on gfx1250.
//   Kernel 1 (parallel):   GI[t][col][b] = (x[b,t,:] @ w_ih^T + b_ih)[col]
//     stored bf16, batch-fastest so kernel 2 reads C-fragments as b128 loads.
//   Kernel 2 (sequential): 4096 dependent [32x256x768] WMMA GEMMs on ONE WGP,
//     r/z-gate weights resident in LDS (bf16), n-gate weights resident in VGPRs,
//     h kept in LDS (bf16 for WMMA A-frags + f32 master copy).
// d_ws requirement: 2048*768*32*2 = 100,663,296 bytes (GI in bf16).
// ---------------------------------------------------------------------------

typedef __attribute__((ext_vector_type(16))) __bf16 v16bf;
typedef __attribute__((ext_vector_type(8)))  __bf16 v8bf;
typedef __attribute__((ext_vector_type(8)))  float  v8f;

#define T_STEPS 2048
#define BATCH   32
#define HID     256
#define G3      768

static __device__ __forceinline__ v16bf cat8(v8bf lo, v8bf hi) {
    return __builtin_shufflevector(lo, hi, 0,1,2,3,4,5,6,7,8,9,10,11,12,13,14,15);
}

// load 8 consecutive f32 (two b128 loads) and convert to 8 bf16
static __device__ __forceinline__ v8bf cvt8(const float* p) {
    float4 a = ((const float4*)p)[0];
    float4 b = ((const float4*)p)[1];
    v8bf r;
    r[0]=(__bf16)a.x; r[1]=(__bf16)a.y; r[2]=(__bf16)a.z; r[3]=(__bf16)a.w;
    r[4]=(__bf16)b.x; r[5]=(__bf16)b.y; r[6]=(__bf16)b.z; r[7]=(__bf16)b.w;
    return r;
}

static __device__ __forceinline__ v8f wmma_bf16(v16bf a, v16bf b, v8f c) {
    return __builtin_amdgcn_wmma_f32_16x16x32_bf16(false, a, false, b,
                                                   (short)0, c, false, false);
}

static __device__ __forceinline__ v8f bf8_to_f32(v8bf x) {
    v8f r;
    #pragma unroll
    for (int j = 0; j < 8; ++j) r[j] = (float)x[j];
    return r;
}

// ---------------------------------------------------------------------------
// Kernel 1: GI = X @ W_ih^T + b_ih over all (t,b) rows.
// GEMM row r = t*32 + b.  Wave task = (mtile, nchunk): 16 rows x 96 cols.
// Output layout: GI[t][col][b]  (batch fastest, bf16).
// ---------------------------------------------------------------------------
__global__ __launch_bounds__(256)
void gi_gemm_kernel(const float* __restrict__ x,     // [B,T,H]
                    const float* __restrict__ w_ih,  // [3H,H]
                    const float* __restrict__ b_ih,  // [3H]
                    __bf16* __restrict__ gi)         // [T,3H,B]
{
    const int tid  = threadIdx.x;
    const int lane = tid & 31;
    const int widx = tid >> 5;
    const int wgl  = blockIdx.x * 8 + widx;
    const int mtile  = wgl >> 3;     // 0..4095
    const int nchunk = wgl & 7;      // 0..7

    const int m  = lane & 15;
    const int h2 = lane >> 4;

    const int t     = mtile >> 1;
    const int bbase = (mtile & 1) << 4;

    const float* xrow  = x + ((long)(bbase + m) * T_STEPS + t) * HID;
    const int    nbase = nchunk * 96;

    v8f acc[6];
    #pragma unroll
    for (int j = 0; j < 6; ++j) {
        const float bias = b_ih[nbase + j*16 + m];
        #pragma unroll
        for (int v = 0; v < 8; ++v) acc[j][v] = bias;
    }

    #pragma unroll
    for (int ks = 0; ks < 8; ++ks) {
        const int k0 = ks*32 + h2*8;
        v16bf afrag = cat8(cvt8(xrow + k0), cvt8(xrow + k0 + 16));
        #pragma unroll
        for (int j = 0; j < 6; ++j) {
            const float* wrow = w_ih + (long)(nbase + j*16 + m) * HID;
            v16bf bfrag = cat8(cvt8(wrow + k0), cvt8(wrow + k0 + 16));
            acc[j] = wmma_bf16(afrag, bfrag, acc[j]);
        }
    }

    // Packed bf16 store: C element v has row bbase+8*h2+v (contiguous in v).
    #pragma unroll
    for (int j = 0; j < 6; ++j) {
        const int col = nbase + j*16 + m;
        v8bf pk;
        #pragma unroll
        for (int v = 0; v < 8; ++v) pk[v] = (__bf16)acc[j][v];
        *(v8bf*)(gi + ((long)t * G3 + col) * BATCH + bbase + h2*8) = pk;
    }
}

// ---------------------------------------------------------------------------
// Kernel 2: sequential recurrence. 1 block, 512 threads (16 waves), one WGP.
// LDS: w_hh r/z gates bf16 (256KB) + h bf16 (16KB) + h f32 (32KB) = 304KB.
// Wave i owns columns [16i,16i+16) of every gate -> elementwise update is local.
// ---------------------------------------------------------------------------
#define LDS_BYTES (262144 + 16384 + 32768)

__global__ __launch_bounds__(512)
void gru_seq_kernel(const __bf16* __restrict__ gi,   // [T,3H,B]
                    const float*  __restrict__ w_hh, // [3H,H]
                    const float*  __restrict__ b_hh, // [3H]
                    const float*  __restrict__ h0,   // [B,H]
                    float* __restrict__ out)         // [T,B,H]
{
    extern __shared__ char smem[];
    __bf16* lds_whh = (__bf16*)smem;                       // rows 0..511 (r,z)
    __bf16* lds_hb  = (__bf16*)(smem + 262144);            // h bf16 [32x256]
    float*  lds_hf  = (float*) (smem + 262144 + 16384);    // h f32  [32x256]

    const int tid  = threadIdx.x;
    const int lane = tid & 31;
    const int widx = tid >> 5;          // 0..15
    const int m    = lane & 15;
    const int h2   = lane >> 4;
    const int col  = widx*16 + m;       // this lane's column in [0,256)

    // ---- one-time init: stage r/z weights in LDS (bf16) -------------------
    for (int i = tid; i < 512*HID/4; i += 512) {
        float4 w = ((const float4*)w_hh)[i];
        __bf16* d = lds_whh + i*4;
        d[0]=(__bf16)w.x; d[1]=(__bf16)w.y; d[2]=(__bf16)w.z; d[3]=(__bf16)w.w;
    }
    for (int i = tid; i < BATCH*HID; i += 512) {
        float hv = h0[i];
        lds_hb[i] = (__bf16)hv;
        lds_hf[i] = hv;
    }

    // ---- one-time init: n-gate weights resident in VGPRs ------------------
    v16bf bn[8];
    {
        const float* wrow = w_hh + (long)(512 + col) * HID;
        #pragma unroll
        for (int ks = 0; ks < 8; ++ks) {
            const int k0 = ks*32 + h2*8;
            bn[ks] = cat8(cvt8(wrow + k0), cvt8(wrow + k0 + 16));
        }
    }
    const float bhr = b_hh[col];
    const float bhz = b_hh[256 + col];
    const float bhn = b_hh[512 + col];

    __syncthreads();

    for (int t = 0; t < T_STEPS; ++t) {
        const __bf16* giT = gi + (long)t * G3 * BATCH;

        // prefetch next step's 48KB gi block (global_prefetch_b8)
        if (t + 1 < T_STEPS) {
            const char* nxt = (const char*)(giT + G3 * BATCH);
            if (tid < 384) __builtin_prefetch(nxt + tid*128, 0, 0);
        }

        // gi fragments for this step: one b128 load each (batch-fastest layout)
        v8f gir[2], giz[2], gin[2];
        #pragma unroll
        for (int mt = 0; mt < 2; ++mt) {
            const int fo = mt*16 + h2*8;
            gir[mt] = bf8_to_f32(*(const v8bf*)(giT + (long)(      col) * BATCH + fo));
            giz[mt] = bf8_to_f32(*(const v8bf*)(giT + (long)(256 + col) * BATCH + fo));
            gin[mt] = bf8_to_f32(*(const v8bf*)(giT + (long)(512 + col) * BATCH + fo));
        }

        #pragma unroll 1
        for (int layer = 0; layer < 2; ++layer) {
            v8f ar[2], az[2], an[2];
            #pragma unroll
            for (int mt = 0; mt < 2; ++mt) {
                ar[mt] = gir[mt];
                az[mt] = giz[mt];
                #pragma unroll
                for (int v = 0; v < 8; ++v) an[mt][v] = 0.0f;
            }

            #pragma unroll
            for (int ks = 0; ks < 8; ++ks) {
                const int k0 = ks*32 + h2*8;
                v16bf a0 = cat8(*(const v8bf*)(lds_hb + ( 0 + m)*HID + k0),
                                *(const v8bf*)(lds_hb + ( 0 + m)*HID + k0 + 16));
                v16bf a1 = cat8(*(const v8bf*)(lds_hb + (16 + m)*HID + k0),
                                *(const v8bf*)(lds_hb + (16 + m)*HID + k0 + 16));
                v16bf br = cat8(*(const v8bf*)(lds_whh + (long)col*HID + k0),
                                *(const v8bf*)(lds_whh + (long)col*HID + k0 + 16));
                v16bf bz = cat8(*(const v8bf*)(lds_whh + (long)(256+col)*HID + k0),
                                *(const v8bf*)(lds_whh + (long)(256+col)*HID + k0 + 16));
                ar[0] = wmma_bf16(a0, br, ar[0]);
                ar[1] = wmma_bf16(a1, br, ar[1]);
                az[0] = wmma_bf16(a0, bz, az[0]);
                az[1] = wmma_bf16(a1, bz, az[1]);
                an[0] = wmma_bf16(a0, bn[ks], an[0]);
                an[1] = wmma_bf16(a1, bn[ks], an[1]);
            }

            // elementwise GRU update (wave-local columns)
            float hnew[2][8];
            #pragma unroll
            for (int mt = 0; mt < 2; ++mt) {
                #pragma unroll
                for (int v = 0; v < 8; ++v) {
                    const int row = mt*16 + h2*8 + v;
                    const float r = 1.0f / (1.0f + __expf(-(ar[mt][v] + bhr)));
                    const float z = 1.0f / (1.0f + __expf(-(az[mt][v] + bhz)));
                    const float n = tanhf(gin[mt][v] + r * (an[mt][v] + bhn));
                    const float ho = lds_hf[row*HID + col];
                    hnew[mt][v] = (1.0f - z) * n + z * ho;
                }
            }

            __syncthreads();   // all waves done reading old h
            #pragma unroll
            for (int mt = 0; mt < 2; ++mt) {
                #pragma unroll
                for (int v = 0; v < 8; ++v) {
                    const int row = mt*16 + h2*8 + v;
                    lds_hf[row*HID + col] = hnew[mt][v];
                    lds_hb[row*HID + col] = (__bf16)hnew[mt][v];
                    if (layer == 1)
                        out[(long)t*BATCH*HID + row*HID + col] = hnew[mt][v];
                }
            }
            __syncthreads();   // new h visible before next GEMM
        }
    }
}

// ---------------------------------------------------------------------------
extern "C" void kernel_launch(void* const* d_in, const int* in_sizes, int n_in,
                              void* d_out, int out_size, void* d_ws, size_t ws_size,
                              hipStream_t stream) {
    // setup_inputs order: batch_size, input, hidden, w_ih, w_hh, b_ih, b_hh
    const float* x      = (const float*)d_in[1];
    const float* hidden = (const float*)d_in[2];
    const float* w_ih   = (const float*)d_in[3];
    const float* w_hh   = (const float*)d_in[4];
    const float* b_ih   = (const float*)d_in[5];
    const float* b_hh   = (const float*)d_in[6];
    __bf16* gi  = (__bf16*)d_ws;      // [T,3H,B] bf16, ~100.7 MB
    float*  out = (float*)d_out;      // [T,B,H] f32

    // (T*B/16) mtiles * 8 nchunks = 32768 wave tasks / 8 waves per block
    gi_gemm_kernel<<<4096, 256, 0, stream>>>(x, w_ih, b_ih, gi);
    gru_seq_kernel<<<1, 512, LDS_BYTES, stream>>>(gi, w_hh, b_hh, hidden, out);
}
